// CRFasRNN_12317966204979
// MI455X (gfx1250) — matrix operand order repaired and checked by reading, not compiled, not run on hardware
//
#include <hip/hip_runtime.h>

// CDNA5 / gfx1250, wave32. Fused CRF-as-RNN mean-field iteration.
// Flash-attention-style: never materializes the 680MB affinity; recomputes
// exp2 tiles on the fly (raw v_exp_f32) and reduces P @ [M | 1] with
// v_wmma_f32_16x16x32_f16. M tile staged TRANSPOSED in LDS so each B operand
// is two ds_load_b128. K-reduction split 2-way across blocks for occupancy,
// combined by a deterministic reduce kernel.

typedef __attribute__((ext_vector_type(16))) _Float16 v16h;
typedef __attribute__((ext_vector_type(2)))  __fp16   pkh2;   // cvt_pkrtz result type
typedef __attribute__((ext_vector_type(8)))  float    v8f;

#define B_SZ      2
#define L_CLS     21
#define NPIX      9216        // 96*96
#define MCH       32          // padded channels of M buffer (21 + denominator + pad)
#define ROWS_WG   128         // 8 waves x 16 rows
#define COLS_TILE 128         // columns staged in LDS per barrier round
#define KSPLIT    2           // column-range split across blocks
#define PCH       24          // padded channels in partial buffer (16B aligned)
#define THREADS   256

#define LOG2E      1.4426950408889634f
#define SQRT_LOG2E 1.2011224087864498f

// Raw v_exp_f32: exponents here are <= 0 and sub-denormal results are
// negligible in the f16 message sum, so skip OCML's denorm-range fixup.
__device__ __forceinline__ float fast_exp2(float x) {
#if __has_builtin(__builtin_amdgcn_exp2f)
    return __builtin_amdgcn_exp2f(x);
#else
    float r;
    asm volatile("v_exp_f32 %0, %1" : "=v"(r) : "v"(x));
    return r;
#endif
}

// ---------------------------------------------------------------------------
// Prep: scaled guide features (f*sqrt(log2e), w = -0.5*log2e*|f|^2);
//       Q0 = softmax(-E0); M0 = MuW @ Q0 stored channel-major [b][32][N] f16.
// ---------------------------------------------------------------------------
__global__ __launch_bounds__(THREADS)
void crf_prep(const float* __restrict__ E0, const float* __restrict__ Refs,
              const float* __restrict__ MuW, float4* __restrict__ fq,
              _Float16* __restrict__ M0)
{
    int idx = blockIdx.x * blockDim.x + threadIdx.x;
    if (idx >= B_SZ * NPIX) return;
    int b = idx / NPIX, n = idx % NPIX;

    const float* rb = Refs + (size_t)b * 3 * NPIX + n;
    float f0 = rb[0], f1 = rb[NPIX], f2 = rb[2 * NPIX];
    float4 v;
    v.x = f0 * SQRT_LOG2E; v.y = f1 * SQRT_LOG2E; v.z = f2 * SQRT_LOG2E;
    v.w = -0.5f * LOG2E * (f0 * f0 + f1 * f1 + f2 * f2);
    fq[idx] = v;

    const float* eb = E0 + (size_t)b * L_CLS * NPIX + n;
    float z[L_CLS], zmax = -3.4e38f;
    #pragma unroll
    for (int l = 0; l < L_CLS; ++l) { z[l] = -eb[(size_t)l * NPIX]; zmax = fmaxf(zmax, z[l]); }
    float s = 0.f;
    #pragma unroll
    for (int l = 0; l < L_CLS; ++l) { float e = __expf(z[l] - zmax); z[l] = e; s += e; }
    float inv = 1.f / s;

    _Float16* mo = M0 + (size_t)b * MCH * NPIX + n;      // channel-major
    #pragma unroll
    for (int k = 0; k < L_CLS; ++k) {
        float mk = 0.f;
        #pragma unroll
        for (int l = 0; l < L_CLS; ++l) mk += MuW[k * L_CLS + l] * (z[l] * inv);
        mo[(size_t)k * NPIX] = (_Float16)mk;
    }
    mo[(size_t)L_CLS * NPIX] = (_Float16)1.0f;           // denominator channel
    #pragma unroll
    for (int k = L_CLS + 1; k < MCH; ++k) mo[(size_t)k * NPIX] = (_Float16)0.0f;
}

// ---------------------------------------------------------------------------
// Partial message: for one row tile (128 rows) and one half of the columns,
//   part[n, j] = sum_m 2^(f_n.f_m + w_n + w_m) * [M(j,m) | 1]
// ---------------------------------------------------------------------------
__global__ __launch_bounds__(THREADS)
void crf_msg_partial(const float4* __restrict__ fq, const _Float16* __restrict__ Min,
                     float* __restrict__ part)
{
    __shared__ float4   fqs[COLS_TILE];                 //  2 KB guide feats of col block
    __shared__ _Float16 Ms[MCH][COLS_TILE];             //  8 KB M tile, TRANSPOSED
    __shared__ float    msgS[ROWS_WG][33];              // 17 KB accumulator spill

    const int tiles = NPIX / ROWS_WG;                   // 72 row tiles
    int b    = blockIdx.x / (KSPLIT * tiles);
    int rest = blockIdx.x % (KSPLIT * tiles);
    int half = rest / tiles;
    int rowb = (rest % tiles) * ROWS_WG;
    int mlo  = half * (NPIX / KSPLIT);
    int mhi  = mlo + (NPIX / KSPLIT);

    int tid  = threadIdx.x;
    int wave = tid >> 5, lane = tid & 31, lh = lane >> 4, r = lane & 15;
    int n    = rowb + wave * 16 + r;                    // A-matrix row for this lane

    const float4*   Fb = fq  + (size_t)b * NPIX;
    const _Float16* Mb = Min + (size_t)b * MCH * NPIX;
    float4 fn = Fb[n];

    v8f acc0 = {};   // channels 0..15
    v8f acc1 = {};   // channels 16..31 (21 = row-sum denominator)

    // staging roles: 8 threads per channel row, 2 uint4 (32B) each
    int sch  = tid >> 3;                                // 0..31 channel
    int sseg = (tid & 7) * 16;                          // column offset, step 16 halves

    for (int m0 = mlo; m0 < mhi; m0 += COLS_TILE) {
        __syncthreads();
        if (tid < COLS_TILE) fqs[tid] = Fb[m0 + tid];
        {
            const uint4* gp = (const uint4*)(Mb + (size_t)sch * NPIX + m0 + sseg);
            uint4* lp = (uint4*)&Ms[sch][sseg];
            lp[0] = gp[0]; lp[1] = gp[1];
            if (m0 + COLS_TILE < mhi) {                 // prefetch next tile
                __builtin_prefetch(Mb + (size_t)sch * NPIX + m0 + COLS_TILE + sseg, 0, 1);
                if (tid < COLS_TILE) __builtin_prefetch(Fb + m0 + COLS_TILE + tid, 0, 1);
            }
        }
        __syncthreads();

        #pragma unroll
        for (int cb = 0; cb < COLS_TILE / 32; ++cb) {
            // ---- A tile (16x32 f16): lane<16 K 0..15, lane>=16 K 16..31
            // Issue all 16 guide loads first (batched ds_load_b128), then compute.
            float4 g[16];
            #pragma unroll
            for (int i = 0; i < 16; ++i) {
                int k = cb * 32 + 16 * lh + i;
                g[i] = fqs[k];
            }
            float pe[16];
            #pragma unroll
            for (int i = 0; i < 16; ++i) {
                float e = fn.w + g[i].w;                // base-2 exponent, <= 0
                e = fmaf(fn.x, g[i].x, e);
                e = fmaf(fn.y, g[i].y, e);
                e = fmaf(fn.z, g[i].z, e);
                pe[i] = fast_exp2(e);
            }
            union AV { v16h v; pkh2 p[8]; } a;
            #pragma unroll
            for (int j = 0; j < 8; ++j)
                a.p[j] = __builtin_amdgcn_cvt_pkrtz(pe[2 * j], pe[2 * j + 1]);

            // ---- B tiles (32x16 f16): col = lane&15 channel, lanes<16 K=0..15, >=16 K=16..31
            union BV { v16h v; uint4 q[2]; } b0, b1;
            const uint4* p0 = (const uint4*)&Ms[r][cb * 32 + 16 * lh];
            const uint4* p1 = (const uint4*)&Ms[16 + r][cb * 32 + 16 * lh];
            b0.q[0] = p0[0]; b0.q[1] = p0[1];
            b1.q[0] = p1[0]; b1.q[1] = p1[1];

            acc0 = __builtin_amdgcn_wmma_f32_16x16x32_f16(false, a.v, false, b0.v,
                                                          (short)0, acc0, false, false);
            acc1 = __builtin_amdgcn_wmma_f32_16x16x32_f16(false, a.v, false, b1.v,
                                                          (short)0, acc1, false, false);
        }
    }

    // Spill C/D tiles (VGPR v -> row v or v+8 per lane half, col = lane&15)
    __syncthreads();
    #pragma unroll
    for (int v = 0; v < 8; ++v) {
        int rr = wave * 16 + v + 8 * lh;
        msgS[rr][r]      = acc0[v];
        msgS[rr][16 + r] = acc1[v];
    }
    __syncthreads();

    if (tid < ROWS_WG) {
        int n2 = rowb + tid;
        float* dst = part + ((size_t)(half * B_SZ + b) * NPIX + n2) * PCH;
        float4* dst4 = (float4*)dst;
        #pragma unroll
        for (int j = 0; j < 6; ++j) {
            float4 t;
            t.x = (4 * j + 0 < 22) ? msgS[tid][4 * j + 0] : 0.f;
            t.y = (4 * j + 1 < 22) ? msgS[tid][4 * j + 1] : 0.f;
            t.z = (4 * j + 2 < 22) ? msgS[tid][4 * j + 2] : 0.f;
            t.w = (4 * j + 3 < 22) ? msgS[tid][4 * j + 3] : 0.f;
            dst4[j] = t;
        }
    }
}

// ---------------------------------------------------------------------------
// Combine the KSPLIT partials (fixed order => deterministic), normalize,
// Q' = softmax(-(E0 + msg)), M' = MuW @ Q' (channel-major f16), Q' -> Qout.
// ---------------------------------------------------------------------------
__global__ __launch_bounds__(128)
void crf_update(const float* __restrict__ E0, const float* __restrict__ MuW,
                const float* __restrict__ part, _Float16* __restrict__ Mout,
                float* __restrict__ Qout)
{
    int idx = blockIdx.x * blockDim.x + threadIdx.x;
    if (idx >= B_SZ * NPIX) return;
    int b = idx / NPIX, n = idx % NPIX;

    const float4* p0 = (const float4*)(part + ((size_t)(0 * B_SZ + b) * NPIX + n) * PCH);
    const float4* p1 = (const float4*)(part + ((size_t)(1 * B_SZ + b) * NPIX + n) * PCH);
    float acc[PCH];
    #pragma unroll
    for (int j = 0; j < 6; ++j) {
        float4 a = p0[j], c = p1[j];
        acc[4 * j + 0] = a.x + c.x; acc[4 * j + 1] = a.y + c.y;
        acc[4 * j + 2] = a.z + c.z; acc[4 * j + 3] = a.w + c.w;
    }
    float invden = 1.0f / acc[L_CLS];

    const float* eb = E0 + (size_t)b * L_CLS * NPIX + n;
    float z[L_CLS], zmax = -3.4e38f;
    #pragma unroll
    for (int l = 0; l < L_CLS; ++l) {
        float zl = -(eb[(size_t)l * NPIX] + acc[l] * invden);
        z[l] = zl; zmax = fmaxf(zmax, zl);
    }
    float s = 0.f;
    #pragma unroll
    for (int l = 0; l < L_CLS; ++l) { float e = __expf(z[l] - zmax); z[l] = e; s += e; }
    float invs = 1.f / s;

    float* qo = Qout + (size_t)b * L_CLS * NPIX + n;
    #pragma unroll
    for (int l = 0; l < L_CLS; ++l) { z[l] *= invs; qo[(size_t)l * NPIX] = z[l]; }

    _Float16* mo = Mout + (size_t)b * MCH * NPIX + n;    // channel-major
    #pragma unroll
    for (int k = 0; k < L_CLS; ++k) {
        float mk = 0.f;
        #pragma unroll
        for (int l = 0; l < L_CLS; ++l) mk += MuW[k * L_CLS + l] * z[l];
        mo[(size_t)k * NPIX] = (_Float16)mk;
    }
    mo[(size_t)L_CLS * NPIX] = (_Float16)1.0f;
    #pragma unroll
    for (int k = L_CLS + 1; k < MCH; ++k) mo[(size_t)k * NPIX] = (_Float16)0.0f;
}

// ---------------------------------------------------------------------------
extern "C" void kernel_launch(void* const* d_in, const int* in_sizes, int n_in,
                              void* d_out, int out_size, void* d_ws, size_t ws_size,
                              hipStream_t stream) {
    const float* E0   = (const float*)d_in[0];   // [2,21,96,96]
    const float* Refs = (const float*)d_in[1];   // [2,3,96,96]
    const float* MuW  = (const float*)d_in[2];   // [21,21]
    float*       out  = (float*)d_out;           // [2,21,96,96]

    char* ws = (char*)d_ws;
    const size_t FQ_BYTES = (size_t)B_SZ * NPIX * sizeof(float4);          // 288 KB
    const size_t M_BYTES  = (size_t)B_SZ * MCH * NPIX * sizeof(_Float16);  // ~1.15 MB
    float4*   fqbuf = (float4*)ws;
    _Float16* M0    = (_Float16*)(ws + FQ_BYTES);
    _Float16* M1    = (_Float16*)(ws + FQ_BYTES + M_BYTES);
    float*    partb = (float*)(ws + FQ_BYTES + 2 * M_BYTES);

    crf_prep<<<(B_SZ * NPIX + THREADS - 1) / THREADS, THREADS, 0, stream>>>(
        E0, Refs, MuW, fqbuf, M0);

    _Float16* mbuf[2] = { M0, M1 };
    const int pblocks = B_SZ * KSPLIT * (NPIX / ROWS_WG);   // 288
    const int ublocks = (B_SZ * NPIX + 127) / 128;          // 144
    for (int it = 0; it < 5; ++it) {
        crf_msg_partial<<<pblocks, THREADS, 0, stream>>>(fqbuf, mbuf[it & 1], partb);
        crf_update<<<ublocks, 128, 0, stream>>>(E0, MuW, partb, mbuf[(it + 1) & 1], out);
    }
}
